// CRFConstituency_4733053960799
// MI455X (gfx1250) — matrix operand order repaired and checked by reading, not compiled, not run on hardware
//
#include <hip/hip_runtime.h>
#include <hip/hip_bf16.h>
#include <math.h>

#define B_DIM 128
#define L_DIM 256
#define LL (L_DIM * L_DIM)
#define SROW 260                                  // row stride: %4==0 for 16B-aligned ds_load_b128
#define NEGV (-4.2535295865117308e37f)            // finfo(f32).min / 8

typedef __attribute__((ext_vector_type(16))) _Float16 v16h;
typedef __attribute__((ext_vector_type(8)))  float    v8f;

// ---------------- kernel 0: zero outputs + accumulators ----------------
__global__ void crf_init_kernel(float* out, float* acc, int n_out) {
  int idx = blockIdx.x * blockDim.x + threadIdx.x;
  int stride = gridDim.x * blockDim.x;
  for (int u = idx; u < n_out; u += stride) out[u] = 0.0f;
  if (idx < 4) acc[idx] = 0.0f;
}

// ---------------- kernel 1: gold score reduction ----------------
__global__ void crf_gold_kernel(const float* __restrict__ scores,
                                const int* __restrict__ lens,
                                const unsigned char* __restrict__ target,
                                float* acc) {
  __shared__ float red[256];
  int idx = blockIdx.x * blockDim.x + threadIdx.x;
  int stride = gridDim.x * blockDim.x;
  float local = 0.f;
  for (int u = idx; u < B_DIM * LL; u += stride) {
    int b  = u >> 16;
    int ij = u & (LL - 1);
    int i  = ij >> 8;
    int j  = ij & 255;
    int len = lens[b];
    bool m = (j > i) && (j <= len) && (target[u] != 0);
    local += m ? scores[u] : 0.f;
  }
  red[threadIdx.x] = local;
  __syncthreads();
  for (int s = 128; s > 0; s >>= 1) {
    if (threadIdx.x < s) red[threadIdx.x] += red[threadIdx.x + s];
    __syncthreads();
  }
  if (threadIdx.x == 0) atomicAdd(&acc[1], red[0]);
}

// ---------------- kernel 2: inside (forward) + outside (backward), one WG per batch ----------------
__global__ __launch_bounds__(256, 1)
void crf_inside_kernel(const float* __restrict__ scores,
                       const int* __restrict__ lens,
                       float* __restrict__ marg,   // B*L*L, pre-zeroed
                       float* acc) {
  // Whole DP table LDS-resident: upper triangle = inside table, lower = gradients.
  __shared__ __align__(16) float s_lds[L_DIM * SROW];   // 266,240 B
  __shared__ float shA[L_DIM];                    // A-side exp shift
  __shared__ float shB[L_DIM];                    // B-side exp shift
  __shared__ float jm[L_DIM];                     // exact joint max per span

  const int b    = blockIdx.x;
  const int tid  = threadIdx.x;
  const int lane = tid & 31;
  const int wv   = tid >> 5;                      // wave id (wave32: 8 waves)
  const float* sc_b = scores + (size_t)b * LL;
  float* marg_b = marg + (size_t)b * LL;
  const int len = lens[b];

  for (int u = tid; u < L_DIM * SROW; u += 256) s_lds[u] = NEGV;
  __syncthreads();
  for (int i = tid; i < L_DIM - 1; i += 256)
    s_lds[i * SROW + i + 1] = sc_b[i * L_DIM + i + 1];
  for (int r = tid; r < L_DIM; r += 256)
    __builtin_prefetch(sc_b + r * L_DIM, 0, 1);   // global_prefetch_b8: warm L2/WGP$
  __syncthreads();

  // -------- forward inside DP: one diagonal (width w) per step --------
  for (int w = 2; w < L_DIM; ++w) {
    const int nvalid = L_DIM - w;

    // phase 1 (VALU): shift statistics for safe f16 exp-space factorization
    if (tid < nvalid) {
      const int i = tid, j = i + w;
      float rm = NEGV, cm = NEGV, jx = NEGV;
      for (int k = i + 1; k < j; ++k) {
        float a = s_lds[i * SROW + k];
        float c = s_lds[k * SROW + j];
        rm = fmaxf(rm, a);
        cm = fmaxf(cm, c);
        jx = fmaxf(jx, a + c);
      }
      float gap = rm + cm - jx;                   // >= 0; split evenly between factors
      float a_shift = rm - 0.5f * gap;
      shA[i] = a_shift;
      shB[i] = jx - a_shift;                      // shA + shB == joint max
      jm[i]  = jx;
    }
    __syncthreads();

    // phase 2 (WMMA): banded logsumexp contraction as f16 matmul in exp space.
    // C = A(16 x 32) x B(32 x 16) accumulated over 32-aligned k-tiles; only diag(C) used.
    const int nblocks = (nvalid + 15) >> 4;
    for (int ib = wv; ib < nblocks; ib += 8) {
      const int i0 = ib << 4;
      v8f cfrag = {};
      int kmax = i0 + 14 + w;
      if (kmax > L_DIM - 2) kmax = L_DIM - 2;     // so k0 <= 224, k0+31 <= 255: no clamping needed

      const int  r      = lane & 15;
      const bool hiHalf = lane >= 16;
      const int  row    = i0 + r;                 // A row (span start i)
      const int  rowc   = row < L_DIM ? row : L_DIM - 1;
      const bool rowOK  = row < nvalid;
      const int  jc     = i0 + r + w;             // B column target j
      const int  jcc    = jc < L_DIM ? jc : L_DIM - 1;
      const bool colOK  = (i0 + r) < nvalid;
      const float sA = rowOK ? shA[row] : 0.f;
      const float sB = colOK ? shB[i0 + r] : 0.f;

      for (int k0 = (i0 + 1) & ~31; k0 <= kmax; k0 += 32) {
        // ---- gather raw values first (branchless, batched under one dscnt wait) ----
        const int abk = k0 + (hiHalf ? 8 : 0);    // this lane's A k-origin
        const int abase = rowc * SROW + abk;      // 16B aligned: SROW%4==0, k0%32==0
        float4 a0 = *(const float4*)&s_lds[abase];
        float4 a1 = *(const float4*)&s_lds[abase + 4];
        float4 a2 = *(const float4*)&s_lds[abase + 16];
        float4 a3 = *(const float4*)&s_lds[abase + 20];
        float av[16] = {a0.x, a0.y, a0.z, a0.w, a1.x, a1.y, a1.z, a1.w,
                        a2.x, a2.y, a2.z, a2.w, a3.x, a3.y, a3.z, a3.w};
        float bvv[16];
#pragma unroll
        for (int h = 0; h < 16; ++h) {
          int kb = k0 + (hiHalf ? 8 : 0) + (h < 8 ? h : h + 8);    // in-bounds by construction
          bvv[h] = s_lds[kb * SROW + jcc];
        }
        // ---- branchless masked exp -> f16 fragments ----
        v16h afrag, bfrag;
#pragma unroll
        for (int h = 0; h < 16; ++h) {
          int ka = abk + (h < 8 ? h : h + 8);
          bool okA = rowOK && (ka > row) && (ka < row + w);
          float xa = okA ? (av[h] - sA) : -1.0e30f;   // exp(-1e30) == 0
          afrag[h] = (_Float16)fminf(__expf(xa), 60000.f);
          int kb = k0 + (hiHalf ? 8 : 0) + (h < 8 ? h : h + 8);
          bool okB = colOK && (kb > i0 + r) && (kb < jc);
          float xb = okB ? (bvv[h] - sB) : -1.0e30f;
          bfrag[h] = (_Float16)fminf(__expf(xb), 60000.f);
        }
        cfrag = __builtin_amdgcn_wmma_f32_16x16x32_f16(
            false, afrag, false, bfrag, (short)0, cfrag, false, false);
      }

      // diag(C): (r,r) r<8 lives in lane r / elem r, r>=8 in lane r+16 / elem r-8
      const bool isLow  = lane < 8;
      const bool isHigh = lane >= 24;
      float d = 0.f;
#pragma unroll
      for (int v = 0; v < 8; ++v) {
        bool sel = (isLow && lane == v) || (isHigh && (lane - 24) == v);
        d = sel ? cfrag[v] : d;
      }
      if (isLow || isHigh) {
        int dr = isLow ? lane : lane - 16;
        int i = i0 + dr;
        if (i < nvalid) {
          int j = i + w;
          d = fmaxf(d, 1e-30f);                   // underflow guard
          s_lds[i * SROW + j] = __logf(d) + jm[i] + sc_b[i * L_DIM + j];
        }
      }
    }
    __syncthreads();
  }

  if (tid == 0) atomicAdd(&acc[0], s_lds[len]);   // logZ contribution: s[0][len]

  // -------- backward (outside): gradients in the lower triangle, ds_add_f32 atomics --------
  for (int u = tid; u < LL; u += 256) {
    int rr = u >> 8, cc = u & 255;
    if (rr > cc) s_lds[rr * SROW + cc] = 0.f;
  }
  __syncthreads();
  if (tid == 0) s_lds[len * SROW + 0] = 1.0f;     // dlogZ/ds[0,len] = 1
  __syncthreads();

  for (int w = L_DIM - 1; w >= 2; --w) {
    const int nvalid = L_DIM - w;
    if (tid < nvalid) {
      const int i = tid, j = i + w;
      float gij = s_lds[j * SROW + i];
      if (gij != 0.f) {
        float total = s_lds[i * SROW + j] - sc_b[i * L_DIM + j];  // the lse value
        for (int k = i + 1; k < j; ++k) {
          float p = __expf(s_lds[i * SROW + k] + s_lds[k * SROW + j] - total);
          float contrib = gij * p;
          atomicAdd(&s_lds[k * SROW + i], contrib);  // g[i,k]
          atomicAdd(&s_lds[j * SROW + k], contrib);  // g[k,j]
        }
      }
    }
    __syncthreads();
  }

  // marginals[i,j] = g[i,j] (zero elsewhere; d_out was pre-zeroed)
  for (int u = tid; u < LL; u += 256) {
    int i = u >> 8, j = u & 255;
    if (j > i) marg_b[u] = s_lds[j * SROW + i];
  }
}

// ---------------- kernel 3: finalize loss ----------------
__global__ void crf_finalize_kernel(const int* __restrict__ lens,
                                    const float* __restrict__ acc,
                                    float* out) {
  __shared__ float red[128];
  int t = threadIdx.x;
  red[t] = (t < B_DIM) ? (float)lens[t] : 0.f;
  __syncthreads();
  for (int s = 64; s > 0; s >>= 1) {
    if (t < s) red[t] += red[t + s];
    __syncthreads();
  }
  if (t == 0) out[0] = (acc[0] - acc[1]) / red[0];
}

extern "C" void kernel_launch(void* const* d_in, const int* in_sizes, int n_in,
                              void* d_out, int out_size, void* d_ws, size_t ws_size,
                              hipStream_t stream) {
  const float* scores = (const float*)d_in[0];
  const int*   lens   = (const int*)d_in[1];
  const unsigned char* target = (const unsigned char*)d_in[2];  // jnp bool -> 1 byte
  float* out = (float*)d_out;                     // [loss, marginals(B*L*L)]
  float* acc = (float*)d_ws;                      // [logZ, gold, pad, pad]

  crf_init_kernel<<<2048, 256, 0, stream>>>(out, acc, out_size);
  crf_gold_kernel<<<1024, 256, 0, stream>>>(scores, lens, target, acc);
  crf_inside_kernel<<<B_DIM, 256, 0, stream>>>(scores, lens, out + 1, acc);
  crf_finalize_kernel<<<1, 128, 0, stream>>>(lens, acc, out);
}
